// Mozafari2018_58772332478594
// MI455X (gfx1250) — matrix operand order repaired and checked by reading, not compiled, not run on hardware
//
#include <hip/hip_runtime.h>

typedef __attribute__((ext_vector_type(16))) _Float16 v16h;
typedef __attribute__((ext_vector_type(8)))  float    v8f;

// ---- workspace layout (in f16 elements) ----
constexpr int PADIN_N  = 15 * 6 * 228 * 228;      // padded f16 input      4,678,560
constexpr int W1F_N    = 2 * 5 * 32 * 16;         // w1 A-fragments            5,120
constexpr int W2F_N    = 15 * 9 * 32 * 16;        // w2 A-fragments           69,120
constexpr int POOL_N   = 15 * 30 * 114 * 114;     // padded pooled spikes  5,848,200
constexpr int PADIN_OFF = 0;
constexpr int W1F_OFF   = PADIN_OFF + PADIN_N;
constexpr int W2F_OFF   = W1F_OFF + W1F_N;
constexpr int POOL_OFF  = W2F_OFF + W2F_N;
constexpr int WS_HALVES = POOL_OFF + POOL_N;      // ~21.2 MB

constexpr int OUT_HALF  = 15 * 240 * 112 * 112;   // 45,158,400 (spk then pot)

__device__ __forceinline__ int klocalA(int grp, int j) {
    // A-fragment k order chosen here (contiguous, B uses the same): lanes 0-15: K=0..15 ; lanes 16-31: K=16..31
    return grp * 16 + j;
}

__device__ __forceinline__ v8f wmma16(v16h a, v16h b, v8f c) {
    return __builtin_amdgcn_wmma_f32_16x16x32_f16(false, a, false, b, (short)0, c, false, false);
}

// ---------------- Kernel 0: prep (pad+cvt input, swizzle weights, zero pool) ----------------
__global__ __launch_bounds__(256) void snn_prep(const float* __restrict__ in,
                                                const float* __restrict__ w1,
                                                const float* __restrict__ w2,
                                                _Float16* __restrict__ ws) {
    int idx = blockIdx.x * 256 + threadIdx.x;
    if (idx >= WS_HALVES) return;
    if (idx < PADIN_N) {
        int x = idx % 228; int t = idx / 228;
        int y = t % 228;   t /= 228;
        int c = t % 6;     int n = t / 6;
        _Float16 v = (_Float16)0.0f;
        if (y >= 2 && y < 226 && x >= 2 && x < 226)
            v = (_Float16)in[((n * 6 + c) * 224 + (y - 2)) * 224 + (x - 2)];
        ws[PADIN_OFF + idx] = v;
    } else if (idx < W1F_OFF + W1F_N) {
        int r = idx - W1F_OFF;
        int j = r & 15, lane = (r >> 4) & 31, kstep = (r >> 9) % 5, mtile = r / (5 * 512);
        int grp = lane >> 4;
        int k  = kstep * 32 + klocalA(grp, j);
        int oc = mtile * 16 + (lane & 15);
        _Float16 v = (_Float16)0.0f;
        if (oc < 30 && k < 150) v = (_Float16)w1[oc * 150 + k];   // zero padding: k>=150 or oc>=30
        ws[idx] = v;
    } else if (idx < W2F_OFF + W2F_N) {
        int r = idx - W2F_OFF;
        int j = r & 15, lane = (r >> 4) & 31, kstep = (r >> 9) % 9, mtile = r / (9 * 512);
        int grp = lane >> 4;
        int k  = kstep * 32 + klocalA(grp, j);
        int oc = mtile * 16 + (lane & 15);
        _Float16 v = (_Float16)0.0f;
        if (k < 270) v = (_Float16)w2[oc * 270 + k];              // zero padding: k>=270
        ws[idx] = v;
    } else {
        ws[idx] = (_Float16)0.0f;   // zero padded pooled-spike buffer (borders stay 0)
    }
}

// ---------------- Kernel 1: conv1 + fire(15) + maxpool2x2 -> padded f16 spikes ----------------
// One wave per 16 pooled pixels, all 30 OCs (2 M-tiles). 4 pool sub-positions accumulated via WMMA.
// Tail-K (150..159) entries of ktab point at offset 0: the A fragment is zero there, so the
// gathered (finite) value contributes exactly 0 -> gathers are branchless.
__global__ __launch_bounds__(256) void snn_conv1_pool(_Float16* __restrict__ ws) {
    __shared__ int ktab[160];               // k -> padded-input offset (0 for padded tail K)
    int tid = threadIdx.x;
    if (tid < 160) {
        int k = tid, off = 0;
        if (k < 150) {
            int ic = k / 25, rr = k % 25, kh = rr / 5, kw = rr % 5;
            off = ic * (228 * 228) + kh * 228 + kw;
        }
        ktab[tid] = off;
    }
    __syncthreads();

    const _Float16* padIn = ws + PADIN_OFF;
    const _Float16* w1f   = ws + W1F_OFF;
    _Float16*       pool  = ws + POOL_OFF;

    int wave = tid >> 5, lane = tid & 31, grp = lane >> 4, nl = lane & 15;
    int ntile = blockIdx.x * 8 + wave;      // 11,760 tiles of 16 pooled pixels
    int p   = ntile * 16 + nl;              // pooled pixel id < 188,160
    int img = p / 12544, r = p % 12544, pr = r / 112, pc = r % 112;
    int base0 = (img * 6 * 228 + 2 * pr) * 228 + 2 * pc;

    v8f acc[2][4];
    v8f zero = {0.f, 0.f, 0.f, 0.f, 0.f, 0.f, 0.f, 0.f};
#pragma unroll
    for (int m = 0; m < 2; ++m)
#pragma unroll
        for (int s = 0; s < 4; ++s) acc[m][s] = zero;

#pragma unroll
    for (int kstep = 0; kstep < 5; ++kstep) {
        v16h a0 = *(const v16h*)(w1f + ((0 * 5 + kstep) * 32 + lane) * 16);
        v16h a1 = *(const v16h*)(w1f + ((1 * 5 + kstep) * 32 + lane) * 16);

        int offs[16];                        // hoisted: same offsets for all 4 sub-positions
#pragma unroll
        for (int j = 0; j < 16; ++j)
            offs[j] = ktab[kstep * 32 + grp * 16 + j];   // B layout: K = 16*grp + j

#pragma unroll
        for (int s = 0; s < 4; ++s) {
            int base = base0 + (s >> 1) * 228 + (s & 1);
            v16h b;
#pragma unroll
            for (int j = 0; j < 16; ++j)
                b[j] = padIn[base + offs[j]];            // branchless gather
            acc[0][s] = wmma16(a0, b, acc[0][s]);
            acc[1][s] = wmma16(a1, b, acc[1][s]);
        }
    }

    // fire (pot > 15) then 2x2 max-pool of binary spikes == OR; write into padded interior
#pragma unroll
    for (int m = 0; m < 2; ++m) {
#pragma unroll
        for (int i = 0; i < 8; ++i) {
            int oc = m * 16 + grp * 8 + i;
            if (oc < 30) {
                float mx = acc[m][0][i];
                mx = fmaxf(mx, acc[m][1][i]);
                mx = fmaxf(mx, acc[m][2][i]);
                mx = fmaxf(mx, acc[m][3][i]);
                _Float16 sp = (mx > 15.0f) ? (_Float16)1.0f : (_Float16)0.0f;
                pool[((img * 30 + oc) * 114 + (pr + 1)) * 114 + (pc + 1)] = sp;
            }
        }
    }
}

// ---------------- Kernel 2: conv2 + fire(10) -> d_out (spk ‖ pot, f32) ----------------
// One block per 16-pixel tile; B fragments staged once in LDS, reused by all 15 M-tiles.
__global__ __launch_bounds__(256) void snn_conv2_fire(const _Float16* __restrict__ ws,
                                                      float* __restrict__ out) {
    __shared__ __align__(32) _Float16 bsh[9][32][16];   // 9 K-steps of B fragments (9 KB)
    const _Float16* pool = ws + POOL_OFF;
    const _Float16* w2f  = ws + W2F_OFF;

    int tid = threadIdx.x;
    int ntile = blockIdx.x;                  // 11,760 tiles

    for (int idx = tid; idx < 9 * 32 * 16; idx += 256) {
        int j = idx & 15, laneb = (idx >> 4) & 31, kstep = idx >> 9;
        int grpb = laneb >> 4, nb = laneb & 15;
        int k = kstep * 32 + grpb * 16 + j;
        int kc = (k < 270) ? k : 269;        // branchless clamp (A is zero for k>=270)
        int p = ntile * 16 + nb;
        int img = p / 12544, rr = p % 12544, pr = rr / 112, pc = rr % 112;
        int ic = kc / 9, r2 = kc % 9, kh = r2 / 3, kw = r2 % 3;
        bsh[kstep][laneb][j] = pool[((img * 30 + ic) * 114 + pr + kh) * 114 + pc + kw];
    }
    __syncthreads();

    int wave = tid >> 5, lane = tid & 31, grp = lane >> 4, nl = lane & 15;
    int p = ntile * 16 + nl;
    int img = p / 12544, r = p % 12544, pr = r / 112, pc = r % 112;
    int pixBase = img * 240 * 12544 + pr * 112 + pc;
    float* spk = out;
    float* pot = out + OUT_HALF;

    for (int mtile = wave; mtile < 15; mtile += 8) {
        v8f acc = {0.f, 0.f, 0.f, 0.f, 0.f, 0.f, 0.f, 0.f};
#pragma unroll
        for (int kstep = 0; kstep < 9; ++kstep) {
            v16h a = *(const v16h*)(w2f + ((mtile * 9 + kstep) * 32 + lane) * 16);
            v16h b = *(const v16h*)(&bsh[kstep][lane][0]);
            acc = wmma16(a, b, acc);
        }
#pragma unroll
        for (int i = 0; i < 8; ++i) {
            int oc = mtile * 16 + grp * 8 + i;
            float pv  = acc[i];
            bool  fire = pv > 10.0f;
            int o = pixBase + oc * 12544;
            spk[o] = fire ? 1.0f : 0.0f;
            pot[o] = fire ? pv   : 0.0f;
        }
    }
}

extern "C" void kernel_launch(void* const* d_in, const int* in_sizes, int n_in,
                              void* d_out, int out_size, void* d_ws, size_t ws_size,
                              hipStream_t stream) {
    const float* in = (const float*)d_in[0];   // (15,6,224,224) f32
    const float* w1 = (const float*)d_in[1];   // (30,6,5,5)     f32
    const float* w2 = (const float*)d_in[2];   // (240,30,3,3)   f32
    _Float16* ws = (_Float16*)d_ws;
    float* out = (float*)d_out;

    int prep_blocks = (WS_HALVES + 255) / 256;
    snn_prep<<<prep_blocks, 256, 0, stream>>>(in, w1, w2, ws);
    snn_conv1_pool<<<1470, 256, 0, stream>>>(ws);          // 11,760 waves / 8 per block
    snn_conv2_fire<<<11760, 256, 0, stream>>>(ws, out);    // one 16-pixel tile per block
}
